// DenseDeepGCN_58282706207002
// MI455X (gfx1250) — matrix operand order repaired and checked by reading, not compiled, not run on hardware
//
#include <hip/hip_runtime.h>
#include <hip/hip_bf16.h>

typedef __attribute__((ext_vector_type(16))) __bf16 v16bf;
typedef __attribute__((ext_vector_type(8)))  float  v8f;

#define NPTS 4096
#define BATCH 4
#define BN (BATCH * NPTS)
#define KNN 16

__device__ __forceinline__ int iclampmax(int v, int hi) { return v < hi ? v : hi; }
__device__ __forceinline__ int iclamp01(int v, int hi) { v = v > 0 ? v : 0; return v < hi ? v : hi; }

__device__ __forceinline__ void cvt8(v16bf& a, int o, float4 q0, float4 q1) {
  a[o+0]=(__bf16)q0.x; a[o+1]=(__bf16)q0.y; a[o+2]=(__bf16)q0.z; a[o+3]=(__bf16)q0.w;
  a[o+4]=(__bf16)q1.x; a[o+5]=(__bf16)q1.y; a[o+6]=(__bf16)q1.z; a[o+7]=(__bf16)q1.w;
}

// ---------------- transpose (B,9,N,1) -> (B*N, 9) ----------------
__global__ void k_transpose(const float* __restrict__ in, float* __restrict__ x) {
  int p = blockIdx.x * blockDim.x + threadIdx.x;
  if (p >= BN) return;
  int b = p / NPTS, n = p % NPTS;
#pragma unroll
  for (int c = 0; c < 9; ++c)
    x[(size_t)p * 9 + c] = in[((size_t)b * 9 + c) * NPTS + n];
}

__global__ void k_fill_zero(unsigned* __restrict__ p, int n) {
  int i = blockIdx.x * blockDim.x + threadIdx.x;
  if (i < n) p[i] = 0u;
}

// ---------------- per-point squared norm of a 64-wide feature slice ----------------
__global__ void k_norm64(const float* __restrict__ F, int ldf, float* __restrict__ c2) {
  int p = blockIdx.x * blockDim.x + threadIdx.x;
  if (p >= BN) return;
  const float* r = F + (size_t)p * ldf;
  float s = 0.f;
#pragma unroll
  for (int c = 0; c < 64; c += 4) {
    float4 q = *(const float4*)(r + c);
    s += q.x * q.x + q.y * q.y + q.z * q.z + q.w * q.w;
  }
  c2[p] = s;
}

// ---------------- head kNN (3-dim xyz): brute-force, LDS-staged ----------------
template<int CDIM, int DIL>
__global__ void k_knn(const float* __restrict__ F, int ldf, int* __restrict__ idxOut) {
  constexpr int KD = KNN * DIL;
  __shared__ float tile[32 * CDIM];
  int tid = threadIdx.x;               // 0..63
  int q = blockIdx.x * 64 + tid;
  int b = q / NPTS;
  int base = b * NPTS;

  float qf[CDIM];
#pragma unroll
  for (int c = 0; c < CDIM; ++c) qf[c] = F[(size_t)q * ldf + c];

  float bd[KD]; int bi[KD];
#pragma unroll
  for (int j = 0; j < KD; ++j) { bd[j] = 3.0e38f; bi[j] = 0; }

  for (int t0 = 0; t0 < NPTS; t0 += 32) {
    __syncthreads();
#if __has_builtin(__builtin_amdgcn_global_load_async_to_lds_b32)
    for (int e = tid; e < 32 * CDIM; e += 64) {
      int cc = e / CDIM, c = e % CDIM;
      __builtin_amdgcn_global_load_async_to_lds_b32(
          (__attribute__((address_space(1))) int*)(F + (size_t)(base + t0 + cc) * ldf + c),
          (__attribute__((address_space(3))) int*)(&tile[e]), 0, 0);
    }
    asm volatile("s_wait_asynccnt 0" ::: "memory");
#else
    for (int e = tid; e < 32 * CDIM; e += 64) {
      int cc = e / CDIM, c = e % CDIM;
      tile[e] = F[(size_t)(base + t0 + cc) * ldf + c];
    }
#endif
    __syncthreads();
    if (t0 + 32 < NPTS)
      __builtin_prefetch(F + (size_t)(base + t0 + 32 + (tid & 31)) * ldf, 0, 1);
    for (int cc = 0; cc < 32; ++cc) {
      float d = 0.f;
      const float* tp = &tile[cc * CDIM];
#pragma unroll
      for (int c = 0; c < CDIM; ++c) { float df = qf[c] - tp[c]; d += df * df; }
      if (d < bd[KD - 1]) {
        int j = KD - 1;
        while (j > 0 && bd[j - 1] > d) { bd[j] = bd[j - 1]; bi[j] = bi[j - 1]; --j; }
        bd[j] = d; bi[j] = t0 + cc;
      }
    }
  }
#pragma unroll
  for (int kk = 0; kk < KNN; ++kk)
    idxOut[(size_t)q * KNN + kk] = bi[kk * DIL];
}

// ---------------- feature-space kNN via WMMA ----------------
// Ranking key = |c|^2 - 2 q.c  (the per-query |q|^2 shift preserves order).
// Block = 4 waves; each wave owns a 16-query tile, A resident in registers.
// Per 16-candidate tile: 2x v_wmma_f32_16x16x32_bf16, keys staged via stride-17
// LDS rows, lanes 0-15 run the ordered top-(16*DIL) insertion for their query.
template<int DIL>
__global__ void k_knn_wmma(const float* __restrict__ F, int ldf,
                           const float* __restrict__ c2,
                           int* __restrict__ idxOut) {
  constexpr int KD = KNN * DIL;
  __shared__ float keys[4][16 * 17];
  int lane = threadIdx.x & 31;
  int wv = threadIdx.x >> 5;
  int l16 = lane & 15, hw = lane >> 4;
  int qbase = blockIdx.x * 64 + wv * 16;
  int q = qbase + l16;                 // this lane's A row (query)
  int base = (qbase / NPTS) * NPTS;    // batch start

  // A tile: 16 queries x 64 dims = 2 K-steps, resident in registers
  v16bf a0, a1;
  {
    const float* xr = F + (size_t)q * ldf;
#pragma unroll
    for (int kt = 0; kt < 2; ++kt) {
      v16bf& a = kt ? a1 : a0;
#pragma unroll
      for (int ch = 0; ch < 2; ++ch) {
        const float* pp = xr + kt * 32 + ch * 16 + hw * 8;
        cvt8(a, ch * 8, *(const float4*)pp, *(const float4*)(pp + 4));
      }
    }
  }

  float bd[KD]; int bi[KD];
#pragma unroll
  for (int j = 0; j < KD; ++j) { bd[j] = 3.0e38f; bi[j] = 0; }

  float* mykeys = &keys[wv][0];
  for (int ct = 0; ct < NPTS / 16; ++ct) {
    int cand = base + ct * 16 + l16;   // this lane's B column (candidate)
    const float* cr = F + (size_t)cand * ldf;
    if (ct + 1 < NPTS / 16)
      __builtin_prefetch(cr + 16 * ldf, 0, 1);
    v8f acc = {};
#pragma unroll
    for (int kt = 0; kt < 2; ++kt) {
      v16bf bm;
      const float* pp = cr + kt * 32 + hw * 8;
#pragma unroll
      for (int e4 = 0; e4 < 2; ++e4)
        cvt8(bm, e4 * 8, *(const float4*)(pp + 16 * e4), *(const float4*)(pp + 16 * e4 + 4));
      acc = __builtin_amdgcn_wmma_f32_16x16x32_bf16(false, kt ? a1 : a0, false, bm,
                                                    (short)0, acc, false, false);
    }
    float c2n = c2[cand];
#pragma unroll
    for (int v = 0; v < 8; ++v)
      mykeys[(v + 8 * hw) * 17 + l16] = c2n - 2.f * acc[v];
    asm volatile("s_wait_dscnt 0" ::: "memory");   // same-wave LDS store->load ordering
    if (lane < 16) {
      const float* krow = &mykeys[lane * 17];
#pragma unroll 4
      for (int cc = 0; cc < 16; ++cc) {
        float d = krow[cc];
        if (d < bd[KD - 1]) {
          int j = KD - 1;
          while (j > 0 && bd[j - 1] > d) { bd[j] = bd[j - 1]; bi[j] = bi[j - 1]; --j; }
          bd[j] = d; bi[j] = ct * 16 + cc;
        }
      }
    }
  }
  if (lane < 16) {
#pragma unroll
    for (int kk = 0; kk < KNN; ++kk)
      idxOut[(size_t)q * KNN + kk] = bi[kk * DIL];
  }
}

// ---------------- EdgeConv via WMMA bf16 ----------------
template<int CIN, bool HASR>
__global__ void k_edgeconv(const float* __restrict__ F, int ldf,
                           const int* __restrict__ idx,
                           const float* __restrict__ W,    // 64 x 2*CIN
                           const float* __restrict__ bias, // 64
                           const float* __restrict__ R, int ldr,
                           float* __restrict__ O, int ldo) {
  constexpr int CIN2 = 2 * CIN;
  constexpr int KT = (CIN2 + 31) / 32;
  int p = blockIdx.x;
  int b = p / NPTS;
  int lane = threadIdx.x & 31;
  int wv = threadIdx.x >> 5;
  int l16 = lane & 15, hw = lane >> 4;

  int j = idx[(size_t)p * KNN + l16];
  const float* xi = F + (size_t)p * ldf;
  const float* xj = F + (size_t)(b * NPTS + j) * ldf;

  int n = wv * 16 + l16;
  const float* wr = W + (size_t)n * CIN2;

  v8f acc = {};
#pragma unroll
  for (int kt = 0; kt < KT; ++kt) {
    const int kb = kt * 32;
    v16bf a, bm;
#pragma unroll
    for (int ch = 0; ch < 2; ++ch) {
      const int cbase = kb + ch * 16;
      if (cbase + 16 <= CIN && (CIN % 4 == 0)) {
        const float* pp = xi + cbase + hw * 8;
        cvt8(a, ch * 8, *(const float4*)pp, *(const float4*)(pp + 4));
      } else if (cbase >= CIN && cbase + 16 <= CIN2 && (CIN % 4 == 0)) {
        const int cb = cbase - CIN + hw * 8;
        float4 qi0 = *(const float4*)(xi + cb), qi1 = *(const float4*)(xi + cb + 4);
        float4 qj0 = *(const float4*)(xj + cb), qj1 = *(const float4*)(xj + cb + 4);
        a[ch*8+0]=(__bf16)(qj0.x-qi0.x); a[ch*8+1]=(__bf16)(qj0.y-qi0.y);
        a[ch*8+2]=(__bf16)(qj0.z-qi0.z); a[ch*8+3]=(__bf16)(qj0.w-qi0.w);
        a[ch*8+4]=(__bf16)(qj1.x-qi1.x); a[ch*8+5]=(__bf16)(qj1.y-qi1.y);
        a[ch*8+6]=(__bf16)(qj1.z-qi1.z); a[ch*8+7]=(__bf16)(qj1.w-qi1.w);
      } else {
#pragma unroll
        for (int e = 0; e < 8; ++e) {
          int c = cbase + hw * 8 + e;
          float vi = xi[iclampmax(c, CIN - 1)];
          int cj = iclamp01(c - CIN, CIN - 1);
          float vd = xj[cj] - xi[cj];
          float v = (c < CIN) ? vi : ((c < CIN2) ? vd : 0.f);
          a[ch*8+e] = (__bf16)v;
        }
      }
    }
    if (kb + 32 <= CIN2 && (CIN2 % 4 == 0)) {
      const float* pp = wr + kb + hw * 8;
#pragma unroll
      for (int e4 = 0; e4 < 2; ++e4)
        cvt8(bm, e4 * 8, *(const float4*)(pp + 16 * e4), *(const float4*)(pp + 16 * e4 + 4));
    } else {
#pragma unroll
      for (int e = 0; e < 16; ++e) {
        int c = kb + hw * 8 + e + (e & 8);
        float wvv = wr[iclampmax(c, CIN2 - 1)];
        bm[e] = (__bf16)((c < CIN2) ? wvv : 0.f);
      }
    }
    acc = __builtin_amdgcn_wmma_f32_16x16x32_bf16(false, a, false, bm,
                                                  (short)0, acc, false, false);
  }

  float bv = bias[n];
  float mval = -3.0e38f;
#pragma unroll
  for (int v = 0; v < 8; ++v)
    mval = fmaxf(mval, fmaxf(acc[v] + bv, 0.f));  // relu before max over k
  mval = fmaxf(mval, __shfl_xor(mval, 16, 32));
  if (hw == 0) {
    float r = HASR ? R[(size_t)p * ldr + n] : 0.f;
    O[(size_t)p * ldo + n] = mval + r;            // residual after max
  }
}

// ---------------- generic Y = act(X @ W^T + b) via WMMA bf16 ----------------
template<int CIN, int CINX, int COUT, bool RELU, bool FMAX>
__global__ void k_gemm(const float* __restrict__ X, int ldx,
                       const float* __restrict__ extra,
                       const float* __restrict__ W,
                       const float* __restrict__ bias,
                       float* __restrict__ Y, int ldy,
                       unsigned* __restrict__ fmax) {
  constexpr int KT = (CIN + 31) / 32;
  int lane = threadIdx.x;
  int l16 = lane & 15, hw = lane >> 4;
  int row0 = blockIdx.x * 16;
  int col0 = (blockIdx.y * blockDim.y + threadIdx.y) * 16;

  int m = row0 + l16;
  const float* xr = X + (size_t)m * ldx;
  float ex = (CIN > CINX) ? extra[m] : 0.f;
  int n = col0 + l16;
  bool nok = (COUT % 16 == 0) || (n < COUT);
  const float* wr = W + (size_t)(nok ? n : 0) * CIN;

  v8f acc = {};
#pragma unroll
  for (int kt = 0; kt < KT; ++kt) {
    const int kb = kt * 32;
    v16bf a, bm;
#pragma unroll
    for (int ch = 0; ch < 2; ++ch) {
      const int cbase = kb + ch * 16;
      if (cbase + 16 <= CINX) {
        const float* pp = xr + cbase + hw * 8;
        cvt8(a, ch * 8, *(const float4*)pp, *(const float4*)(pp + 4));
      } else {
#pragma unroll
        for (int e = 0; e < 8; ++e) {
          int c = cbase + hw * 8 + e;
          a[ch*8+e] = (__bf16)((c < CIN) ? ex : 0.f);
        }
      }
    }
    if (kb + 32 <= CIN && (CIN % 4 == 0)) {
      const float* pp = wr + kb + hw * 8;
#pragma unroll
      for (int e4 = 0; e4 < 4; ++e4) {
        float4 qq = *(const float4*)(pp + 16 * (e4 >> 1) + 4 * (e4 & 1));
        bm[4*e4+0]=(__bf16)(nok ? qq.x : 0.f); bm[4*e4+1]=(__bf16)(nok ? qq.y : 0.f);
        bm[4*e4+2]=(__bf16)(nok ? qq.z : 0.f); bm[4*e4+3]=(__bf16)(nok ? qq.w : 0.f);
      }
    } else {
#pragma unroll
      for (int e = 0; e < 16; ++e) {
        int c = kb + hw * 8 + e + (e & 8);
        float wvv = wr[iclampmax(c, CIN - 1)];
        bm[e] = (__bf16)((nok && c < CIN) ? wvv : 0.f);
      }
    }
    acc = __builtin_amdgcn_wmma_f32_16x16x32_bf16(false, a, false, bm,
                                                  (short)0, acc, false, false);
  }

  float bv = nok ? bias[n] : 0.f;
#pragma unroll
  for (int v = 0; v < 8; ++v) {
    int mr = row0 + v + 8 * hw;
    float val = acc[v] + bv;
    if (RELU) val = fmaxf(val, 0.f);
    if (nok) {
      if (FMAX) atomicMax(&fmax[mr], __float_as_uint(val));  // relu out >= 0 -> bit max ok
      else      Y[(size_t)mr * ldy + n] = val;
    }
  }
}

// ---------------- log_softmax over 13 classes + transpose to (B,13,N) ----------------
__global__ void k_logsoftmax(const float* __restrict__ logits, float* __restrict__ out) {
  int q = blockIdx.x * blockDim.x + threadIdx.x;
  if (q >= BN) return;
  int b = q / NPTS, n = q % NPTS;
  const float* lr = logits + (size_t)q * 13;
  float m = lr[0];
#pragma unroll
  for (int c = 1; c < 13; ++c) m = fmaxf(m, lr[c]);
  float s = 0.f;
#pragma unroll
  for (int c = 0; c < 13; ++c) s += __expf(lr[c] - m);
  float lse = __logf(s) + m;
#pragma unroll
  for (int c = 0; c < 13; ++c)
    out[((size_t)b * 13 + c) * NPTS + n] = lr[c] - lse;
}

extern "C" void kernel_launch(void* const* d_in, const int* in_sizes, int n_in,
                              void* d_out, int out_size, void* d_ws, size_t ws_size,
                              hipStream_t stream) {
  const float* inputs   = (const float*)d_in[0];
  const float* W_head   = (const float*)d_in[1];
  const float* b_head   = (const float*)d_in[2];
  const float* W_blocks = (const float*)d_in[3];
  const float* b_blocks = (const float*)d_in[4];
  const float* W_fus    = (const float*)d_in[5];
  const float* b_fus    = (const float*)d_in[6];
  const float* W_p1     = (const float*)d_in[7];
  const float* b_p1     = (const float*)d_in[8];
  const float* W_p2     = (const float*)d_in[9];
  const float* b_p2     = (const float*)d_in[10];
  const float* W_p3     = (const float*)d_in[11];
  const float* b_p3     = (const float*)d_in[12];

  char* ws = (char*)d_ws;
  float* x      = (float*)ws; ws += (size_t)BN * 9 * 4;
  float* feats  = (float*)ws; ws += (size_t)BN * 448 * 4;   // 7 blocks of 64, row stride 448
  int*   idx    = (int*)ws;   ws += (size_t)BN * 16 * 4;
  float* fusion = (float*)ws; ws += (size_t)BN * 4;
  float* norms  = (float*)ws; ws += (size_t)BN * 4;
  float* h1     = (float*)ws; ws += (size_t)BN * 512 * 4;
  float* h2     = (float*)ws; ws += (size_t)BN * 256 * 4;
  float* logit  = (float*)ws; ws += (size_t)BN * 13 * 4;

  k_transpose<<<BN / 256, 256, 0, stream>>>(inputs, x);

  // head: kNN on xyz (first 3 of 9 dims), EdgeConv 18 -> 64
  k_knn<3, 1><<<BN / 64, 64, 0, stream>>>(x, 9, idx);
  k_edgeconv<9, false><<<BN, 128, 0, stream>>>(x, 9, idx, W_head, b_head,
                                               nullptr, 0, feats, 448);

  // 6 residual blocks: WMMA kNN in 64-d feature space, dilation i+1
  for (int i = 0; i < 6; ++i) {
    const float* fin = feats + i * 64;
    k_norm64<<<BN / 256, 256, 0, stream>>>(fin, 448, norms);
    switch (i + 1) {
      case 1: k_knn_wmma<1><<<BN / 64, 128, 0, stream>>>(fin, 448, norms, idx); break;
      case 2: k_knn_wmma<2><<<BN / 64, 128, 0, stream>>>(fin, 448, norms, idx); break;
      case 3: k_knn_wmma<3><<<BN / 64, 128, 0, stream>>>(fin, 448, norms, idx); break;
      case 4: k_knn_wmma<4><<<BN / 64, 128, 0, stream>>>(fin, 448, norms, idx); break;
      case 5: k_knn_wmma<5><<<BN / 64, 128, 0, stream>>>(fin, 448, norms, idx); break;
      default: k_knn_wmma<6><<<BN / 64, 128, 0, stream>>>(fin, 448, norms, idx); break;
    }
    k_edgeconv<64, true><<<BN, 128, 0, stream>>>(fin, 448, idx,
                                                 W_blocks + (size_t)i * 64 * 128,
                                                 b_blocks + i * 64,
                                                 fin, 448,                    // residual
                                                 feats + (i + 1) * 64, 448);  // output column
  }

  k_fill_zero<<<BN / 256, 256, 0, stream>>>((unsigned*)fusion, BN);

  dim3 blk(32, 4);
  // fusion: 448 -> 1024, relu, channel-max folded into epilogue via atomicMax
  k_gemm<448, 448, 1024, true, true><<<dim3(BN / 16, 1024 / 64), blk, 0, stream>>>(
      feats, 448, nullptr, W_fus, b_fus, nullptr, 0, (unsigned*)fusion);
  // p1: concat(feats, fusion) = 449 -> 512, relu
  k_gemm<449, 448, 512, true, false><<<dim3(BN / 16, 512 / 64), blk, 0, stream>>>(
      feats, 448, fusion, W_p1, b_p1, h1, 512, nullptr);
  // p2: 512 -> 256, relu
  k_gemm<512, 512, 256, true, false><<<dim3(BN / 16, 256 / 64), blk, 0, stream>>>(
      h1, 512, nullptr, W_p2, b_p2, h2, 256, nullptr);
  // p3: 256 -> 13, no relu
  k_gemm<256, 256, 13, false, false><<<dim3(BN / 16, 1), blk, 0, stream>>>(
      h2, 256, nullptr, W_p3, b_p3, logit, 13, nullptr);

  k_logsoftmax<<<BN / 256, 256, 0, stream>>>(logit, (float*)d_out);
}
